// PCD_Align_51049981281490
// MI455X (gfx1250) — matrix operand order repaired and checked by reading, not compiled, not run on hardware
//
#include <hip/hip_runtime.h>
#include <stdint.h>

#define NF 64
#define DG 8
#define BATCH 2

// ---------------- WMMA types (CDNA5 gfx1250, wave32) ----------------
typedef __attribute__((ext_vector_type(16))) __bf16 v16bf;
typedef __attribute__((ext_vector_type(8)))  float  v8f;

union Frag16 { uint4 q[2]; v16bf v; };

__device__ __forceinline__ unsigned short f2bf(float f) {
    unsigned int u = __float_as_uint(f);
    u += 0x7FFFu + ((u >> 16) & 1u);      // round-to-nearest-even
    return (unsigned short)(u >> 16);
}

// async copy of 16 bytes global -> LDS (CDNA5, tracked by ASYNCcnt)
__device__ __forceinline__ void async_copy16(unsigned int ldsAddr, const void* gptr) {
    unsigned long long ga = (unsigned long long)(uintptr_t)gptr;
    asm volatile("global_load_async_to_lds_b128 %0, %1, off"
                 :: "v"(ldsAddr), "v"(ga) : "memory");
}
__device__ __forceinline__ void async_wait0() {
    asm volatile("s_wait_asynccnt 0x0" ::: "memory");
}
__device__ __forceinline__ unsigned int lds_addr_of(const void* p) {
    // generic pointers to LDS carry the LDS byte offset in the low 32 bits
    return (unsigned int)(uintptr_t)p;
}

// ============================================================================
// Weight repack: fp32 [Co,Ci,3,3] -> bf16 [CoPad][tap*Ci + ci]  (tap-major K)
// ============================================================================
__global__ void repack_w_kernel(const float* __restrict__ w, unsigned short* __restrict__ out,
                                int Co, int Ci, int CoPad) {
    int Ktot = Ci * 9;
    long long idx = (long long)blockIdx.x * blockDim.x + threadIdx.x;
    long long total = (long long)CoPad * Ktot;
    if (idx >= total) return;
    int co  = (int)(idx / Ktot);
    int k   = (int)(idx % Ktot);
    int tap = k / Ci;
    int ci  = k % Ci;
    float v = 0.f;
    if (co < Co) v = w[((long long)co * Ci + ci) * 9 + tap];
    out[idx] = f2bf(v);
}

// ============================================================================
// Implicit-GEMM 3x3 conv (stride 1, pad 1), bf16 WMMA, f32 accumulate.
// Input = one or two 64-ch NCHW fp32 tensors (concat on the fly).
// Block = 256 threads (8 wave32); tile = 64 out-ch x 32 pixels.
// Weights (64-row slab) async-copied to LDS; halo tile staged as bf16.
// ============================================================================
template <int CIN>
__global__ __launch_bounds__(256)
void conv3x3_wmma_kernel(const float* __restrict__ xA, const float* __restrict__ xB,
                         const unsigned short* __restrict__ wpack,
                         const float* __restrict__ bias,
                         float* __restrict__ out,
                         int Cout, int outCAlloc,
                         int H, int W, int act) {
    constexpr int KTOT = CIN * 9;
    extern __shared__ unsigned short smem[];
    unsigned short* sIn = smem;                       // [3][34][CIN] bf16
    unsigned short* sW  = smem + 3 * 34 * CIN;        // [64][KTOT]   bf16

    const int tilesX = W >> 5;
    int bidx = blockIdx.x;
    int xseg = bidx % tilesX;
    int t    = bidx / tilesX;
    int y    = t % H;
    int b    = t / H;
    int tileX = xseg << 5;
    int tid  = threadIdx.x;

    // ---- async copy of this block's 64-row weight slab into LDS
    {
        const char* wsrc = (const char*)(wpack + (size_t)(blockIdx.y * 64) * KTOT);
        unsigned int ldsBase = lds_addr_of(sW);
        constexpr int BYTES = 64 * KTOT * 2;
#pragma unroll 4
        for (int off = tid * 16; off < BYTES; off += 256 * 16)
            async_copy16(ldsBase + off, wsrc + off);
    }

    // ---- stage 3x34xCIN halo tile into LDS as bf16 (coalesced: col innermost)
    for (int i = tid; i < 3 * 34 * CIN; i += 256) {
        int c  = i % 34;
        int t2 = i / 34;
        int ci = t2 % CIN;
        int r  = t2 / CIN;
        int gy = y + r - 1;
        int gx = tileX + c - 1;
        float v = 0.f;
        if (gy >= 0 && gy < H && gx >= 0 && gx < W) {
            const float* src = (CIN == 128 && ci >= 64) ? xB : xA;
            int cc = ci & 63;
            v = src[(((size_t)b * 64 + cc) * H + gy) * W + gx];
        }
        sIn[(r * 34 + c) * CIN + ci] = f2bf(v);
    }

    async_wait0();
    __syncthreads();

    int lane  = tid & 31;
    int wv    = tid >> 5;
    int mtile = wv >> 1;          // 0..3  -> 16 out-channels each
    int pg    = wv & 1;           // 0..1  -> 16 pixels each
    int lhalf = (lane >> 4) & 1;  // lane group 0-15 / 16-31
    int l15   = lane & 15;
    int n     = pg * 16 + l15;    // pixel column within tile

    const unsigned short* wrow = sW + (mtile * 16 + l15) * KTOT;
    const unsigned short* brow = sIn + n * CIN;

    v8f acc = {0.f, 0.f, 0.f, 0.f, 0.f, 0.f, 0.f, 0.f};

#pragma unroll
    for (int kt = 0; kt < KTOT; kt += 32) {
        Frag16 fa, fb;
        // A 16x32 bf16: lanes 0-15 hold K 0..15, lanes 16-31 K 16..31
        const unsigned short* ap = wrow + kt + lhalf * 16;
        fa.q[0] = *(const uint4*)(ap);
        fa.q[1] = *(const uint4*)(ap + 8);

        // B 32x16 bf16: lanes 0-15 K 0..15, lanes 16-31 K 16..31 (col = lane&15)
        int kb  = kt + lhalf * 16;          // 16 consecutive k, single tap (16|CIN)
        int tap = kb / CIN;                 // compile-time folded (kt constant)
        int ci0 = kb % CIN;
        int di = tap / 3, dj = tap % 3;
        const unsigned short* bp = brow + (di * 34 + dj) * CIN + ci0;
        fb.q[0] = *(const uint4*)(bp);
        fb.q[1] = *(const uint4*)(bp + 8);

        acc = __builtin_amdgcn_wmma_f32_16x16x32_bf16(
            false, fa.v, false, fb.v, (short)0, acc, false, false);
    }

    // ---- epilogue: bias (+ LeakyReLU 0.1), fp32 NCHW store (coalesced in x)
    int x = tileX + n;
#pragma unroll
    for (int r = 0; r < 8; ++r) {
        int co = blockIdx.y * 64 + mtile * 16 + r + lhalf * 8;   // C/D row layout
        if (co < Cout) {
            float v = acc[r] + bias[co];
            if (act) v = (v >= 0.f) ? v : 0.1f * v;
            out[(((size_t)b * outCAlloc + co) * H + y) * W + x] = v;
        }
    }
}

// ============================================================================
// Bilinear 2x upsample (half-pixel centers, edge clamp), optional scale.
// ============================================================================
__global__ void upsample2x_kernel(const float* __restrict__ in, float* __restrict__ out,
                                  int h, int w, float scale) {
    int H = h * 2, W = w * 2;
    size_t total = (size_t)BATCH * 64 * H * W;
    size_t idx = (size_t)blockIdx.x * 256 + threadIdx.x;
    if (idx >= total) return;
    int x = (int)(idx % W);
    size_t t = idx / W;
    int y = (int)(t % H);
    int bc = (int)(t / H);
    const float* src = in + (size_t)bc * h * w;
    float sy = (y + 0.5f) * 0.5f - 0.5f;
    float sx = (x + 0.5f) * 0.5f - 0.5f;
    float fy = floorf(sy), fx = floorf(sx);
    int y0 = (int)fy, x0 = (int)fx;
    float ly = sy - fy, lx = sx - fx;
    int y0c = y0 < 0 ? 0 : (y0 >= h ? h - 1 : y0);
    int y1c = (y0 + 1) >= h ? h - 1 : (y0 + 1 < 0 ? 0 : y0 + 1);
    int x0c = x0 < 0 ? 0 : (x0 >= w ? w - 1 : x0);
    int x1c = (x0 + 1) >= w ? w - 1 : (x0 + 1 < 0 ? 0 : x0 + 1);
    float v00 = src[y0c * w + x0c], v01 = src[y0c * w + x1c];
    float v10 = src[y1c * w + x0c], v11 = src[y1c * w + x1c];
    float v = (1.f - ly) * ((1.f - lx) * v00 + lx * v01)
            +        ly  * ((1.f - lx) * v10 + lx * v11);
    out[idx] = v * scale;
}

// ============================================================================
// Modulated deformable conv (DCNv2): gather+modulate into LDS (bf16),
// then per-tap GEMM via WMMA (K = tap*64 + ci). Weights async-staged to LDS.
// ============================================================================
__global__ __launch_bounds__(256)
void dcn_wmma_kernel(const float* __restrict__ xin, const float* __restrict__ om,
                     const unsigned short* __restrict__ wpack,
                     const float* __restrict__ bias,
                     float* __restrict__ out,
                     int H, int W, int act) {
    extern __shared__ unsigned short smem[];
    unsigned short* sSamp = smem;            // [9][32][64] bf16 = 18432
    unsigned short* sW    = smem + 18432;    // [64][576]   bf16 = 36864

    const int OMC = 216;
    const int tilesX = W >> 5;
    int bidx = blockIdx.x;
    int xseg = bidx % tilesX;
    int t    = bidx / tilesX;
    int y    = t % H;
    int b    = t / H;
    int tileX = xseg << 5;
    int tid  = threadIdx.x;

    // ---- async weight slab (64 x 576 bf16 = 73728 B) -> LDS
    {
        unsigned int ldsBase = lds_addr_of(sW);
#pragma unroll 4
        for (int off = tid * 16; off < 64 * 576 * 2; off += 256 * 16)
            async_copy16(ldsBase + off, (const char*)wpack + off);
    }

    // ---- phase 1: deformable bilinear sampling * sigmoid(mask) -> LDS bf16
    for (int idx = tid; idx < DG * 9 * 32; idx += 256) {
        int n   = idx & 31;
        int t2  = idx >> 5;
        int tap = t2 % 9;
        int g   = t2 / 9;
        int x   = tileX + n;

        size_t plane = (size_t)H * W;
        size_t obase = ((size_t)b * OMC) * plane + (size_t)y * W + x;
        int cOff = (g * 9 + tap) * 2;
        float offY = om[obase + (size_t)cOff * plane];
        float offX = om[obase + (size_t)(cOff + 1) * plane];
        float mRaw = om[obase + (size_t)(144 + g * 9 + tap) * plane];
        float mask = 1.0f / (1.0f + __expf(-mRaw));

        float py = (float)y + (float)(tap / 3 - 1) + offY;
        float px = (float)x + (float)(tap % 3 - 1) + offX;
        float fy = floorf(py), fx = floorf(px);
        float ly = py - fy, lx = px - fx;
        int iy0 = (int)fy, ix0 = (int)fx;
        int iy1 = iy0 + 1, ix1 = ix0 + 1;
        bool vy0 = (iy0 >= 0) && (iy0 < H);
        bool vy1 = (iy1 >= 0) && (iy1 < H);
        bool vx0 = (ix0 >= 0) && (ix0 < W);
        bool vx1 = (ix1 >= 0) && (ix1 < W);
        float w00 = (vy0 && vx0) ? (1.f - ly) * (1.f - lx) * mask : 0.f;
        float w01 = (vy0 && vx1) ? (1.f - ly) * lx * mask : 0.f;
        float w10 = (vy1 && vx0) ? ly * (1.f - lx) * mask : 0.f;
        float w11 = (vy1 && vx1) ? ly * lx * mask : 0.f;
        int cy0 = iy0 < 0 ? 0 : (iy0 >= H ? H - 1 : iy0);
        int cy1 = iy1 < 0 ? 0 : (iy1 >= H ? H - 1 : iy1);
        int cx0 = ix0 < 0 ? 0 : (ix0 >= W ? W - 1 : ix0);
        int cx1 = ix1 < 0 ? 0 : (ix1 >= W ? W - 1 : ix1);
        int o00 = cy0 * W + cx0, o01 = cy0 * W + cx1;
        int o10 = cy1 * W + cx0, o11 = cy1 * W + cx1;

#pragma unroll
        for (int cg = 0; cg < 8; ++cg) {
            int c = g * 8 + cg;
            const float* xb = xin + ((size_t)b * 64 + c) * plane;
            float v = w00 * xb[o00] + w01 * xb[o01] + w10 * xb[o10] + w11 * xb[o11];
            sSamp[(tap * 32 + n) * 64 + c] = f2bf(v);
        }
    }

    async_wait0();
    __syncthreads();

    // ---- phase 2: GEMM, Ktot = 9*64 = 576, fully unrolled
    int lane  = tid & 31;
    int wv    = tid >> 5;
    int mtile = wv >> 1;
    int pg    = wv & 1;
    int lhalf = (lane >> 4) & 1;
    int l15   = lane & 15;
    int n     = pg * 16 + l15;

    const unsigned short* wrow = sW + (mtile * 16 + l15) * 576;
    const unsigned short* brow = sSamp + n * 64;

    v8f acc = {0.f, 0.f, 0.f, 0.f, 0.f, 0.f, 0.f, 0.f};

#pragma unroll
    for (int kt = 0; kt < 576; kt += 32) {
        Frag16 fa, fb;
        const unsigned short* ap = wrow + kt + lhalf * 16;
        fa.q[0] = *(const uint4*)(ap);
        fa.q[1] = *(const uint4*)(ap + 8);

        int kb  = kt + lhalf * 16;
        int tap = kb >> 6;
        int ci0 = kb & 63;
        const unsigned short* bp = brow + tap * (32 * 64) + ci0;
        fb.q[0] = *(const uint4*)(bp);
        fb.q[1] = *(const uint4*)(bp + 8);

        acc = __builtin_amdgcn_wmma_f32_16x16x32_bf16(
            false, fa.v, false, fb.v, (short)0, acc, false, false);
    }

    int x = tileX + n;
#pragma unroll
    for (int r = 0; r < 8; ++r) {
        int co = mtile * 16 + r + lhalf * 8;
        float v = acc[r] + bias[co];
        if (act) v = (v >= 0.f) ? v : 0.1f * v;
        out[(((size_t)b * 64 + co) * H + y) * W + x] = v;
    }
}

// ============================================================================
// Host orchestration
// ============================================================================
extern "C" void kernel_launch(void* const* d_in, const int* in_sizes, int n_in,
                              void* d_out, int out_size, void* d_ws, size_t ws_size,
                              hipStream_t stream) {
    (void)in_sizes; (void)n_in; (void)out_size; (void)ws_size;

    const int SM64  = (3 * 34 * 64 + 64 * 576) * 2;     //  86784 B
    const int SM128 = (3 * 34 * 128 + 64 * 1152) * 2;   // 173568 B
    const int SMDCN = (18432 + 36864) * 2;              // 110592 B
    hipFuncSetAttribute(reinterpret_cast<const void*>(conv3x3_wmma_kernel<64>),
                        hipFuncAttributeMaxDynamicSharedMemorySize, SM64);
    hipFuncSetAttribute(reinterpret_cast<const void*>(conv3x3_wmma_kernel<128>),
                        hipFuncAttributeMaxDynamicSharedMemorySize, SM128);
    hipFuncSetAttribute(reinterpret_cast<const void*>(dcn_wmma_kernel),
                        hipFuncAttributeMaxDynamicSharedMemorySize, SMDCN);

    // features (setup_inputs dict order)
    const float* f1_1 = (const float*)d_in[0];
    const float* f1_2 = (const float*)d_in[1];
    const float* f1_3 = (const float*)d_in[2];
    const float* f2_1 = (const float*)d_in[3];
    const float* f2_2 = (const float*)d_in[4];
    const float* f2_3 = (const float*)d_in[5];

    // params: jax pytree flatten (sorted keys at every level)
    const float* L1_dcn_b    = (const float*)d_in[6];
    const float* L1_dcn_omb  = (const float*)d_in[7];
    const float* L1_dcn_omw  = (const float*)d_in[8];
    const float* L1_dcn_w    = (const float*)d_in[9];
    const float* L1_fea_b    = (const float*)d_in[10];
    const float* L1_fea_w    = (const float*)d_in[11];
    const float* L1_oc1_b    = (const float*)d_in[12];
    const float* L1_oc1_w    = (const float*)d_in[13];
    const float* L1_oc2_b1   = (const float*)d_in[14];
    const float* L1_oc2_b2   = (const float*)d_in[15];
    const float* L1_oc2_w1   = (const float*)d_in[16];
    const float* L1_oc2_w2   = (const float*)d_in[17];
    const float* L2_dcn_b    = (const float*)d_in[18];
    const float* L2_dcn_omb  = (const float*)d_in[19];
    const float* L2_dcn_omw  = (const float*)d_in[20];
    const float* L2_dcn_w    = (const float*)d_in[21];
    const float* L2_fea_b    = (const float*)d_in[22];
    const float* L2_fea_w    = (const float*)d_in[23];
    const float* L2_oc1_b    = (const float*)d_in[24];
    const float* L2_oc1_w    = (const float*)d_in[25];
    const float* L2_oc2_b1   = (const float*)d_in[26];
    const float* L2_oc2_b2   = (const float*)d_in[27];
    const float* L2_oc2_w1   = (const float*)d_in[28];
    const float* L2_oc2_w2   = (const float*)d_in[29];
    const float* L3_dcn_b    = (const float*)d_in[30];
    const float* L3_dcn_omb  = (const float*)d_in[31];
    const float* L3_dcn_omw  = (const float*)d_in[32];
    const float* L3_dcn_w    = (const float*)d_in[33];
    const float* L3_oc_b1    = (const float*)d_in[34];
    const float* L3_oc_b2    = (const float*)d_in[35];
    const float* L3_oc_w1    = (const float*)d_in[36];
    const float* L3_oc_w2    = (const float*)d_in[37];
    const float* cas_dcn_b   = (const float*)d_in[38];
    const float* cas_dcn_omb = (const float*)d_in[39];
    const float* cas_dcn_omw = (const float*)d_in[40];
    const float* cas_dcn_w   = (const float*)d_in[41];
    const float* cas_oc_b1   = (const float*)d_in[42];
    const float* cas_oc_b2   = (const float*)d_in[43];
    const float* cas_oc_w1   = (const float*)d_in[44];
    const float* cas_oc_w2   = (const float*)d_in[45];

    // outputs (tuple concat): l1f, l1o, l2o, l3o, off
    const size_t N1 = (size_t)BATCH * 64 * 256 * 256;
    const size_t N2 = (size_t)BATCH * 64 * 128 * 128;
    const size_t N3 = (size_t)BATCH * 64 * 64 * 64;
    float* o_l1f = (float*)d_out;
    float* o_l1o = o_l1f + N1;
    float* o_l2o = o_l1o + N1;
    float* o_l3o = o_l2o + N2;
    float* o_off = o_l3o + N3;

    // workspace allocator
    char* wsBase = (char*)d_ws;
    size_t wsOff = 0;
    auto alloc = [&](size_t bytes) -> void* {
        void* p = wsBase + wsOff;
        wsOff = (wsOff + bytes + 255) & ~(size_t)255;
        return p;
    };
    auto allocPack = [&](int CoPad, int Ktot) {
        return (unsigned short*)alloc((size_t)CoPad * Ktot * 2);
    };
    auto allocF = [&](size_t nElem) { return (float*)alloc(nElem * 4); };

    // packed weights (bf16, tap-major K)
    unsigned short* p_L3oc1  = allocPack(64, 1152);
    unsigned short* p_L3oc2  = allocPack(64, 576);
    unsigned short* p_L3om   = allocPack(256, 576);
    unsigned short* p_L3w    = allocPack(64, 576);
    unsigned short* p_L2oc1  = allocPack(64, 1152);
    unsigned short* p_L2oc2a = allocPack(64, 1152);
    unsigned short* p_L2oc2b = allocPack(64, 576);
    unsigned short* p_L2om   = allocPack(256, 576);
    unsigned short* p_L2w    = allocPack(64, 576);
    unsigned short* p_L2fea  = allocPack(64, 1152);
    unsigned short* p_L1oc1  = allocPack(64, 1152);
    unsigned short* p_L1oc2a = allocPack(64, 1152);
    unsigned short* p_L1oc2b = allocPack(64, 576);
    unsigned short* p_L1om   = allocPack(256, 576);
    unsigned short* p_L1w    = allocPack(64, 576);
    unsigned short* p_L1fea  = allocPack(64, 1152);
    unsigned short* p_casoc1 = allocPack(64, 1152);
    unsigned short* p_casoc2 = allocPack(64, 576);
    unsigned short* p_casom  = allocPack(256, 576);
    unsigned short* p_casw   = allocPack(64, 576);

    // activation scratch
    float* t3a = allocF(N3);
    float* l3f = allocF(N3);
    float* A2  = allocF(N2);
    float* B2b = allocF(N2);
    float* C2  = allocF(N2);
    float* D2  = allocF(N2);
    float* A1  = allocF(N1);
    float* B1b = allocF(N1);
    float* C1  = allocF(N1);
    float* E1  = allocF(N1);
    float* up  = allocF(N1);
    float* om  = allocF((size_t)BATCH * 216 * 256 * 256);

    auto repack = [&](const float* w, unsigned short* dst, int Co, int Ci, int CoPad) {
        long long total = (long long)CoPad * Ci * 9;
        int blocks = (int)((total + 255) / 256);
        repack_w_kernel<<<blocks, 256, 0, stream>>>(w, dst, Co, Ci, CoPad);
    };
    auto conv = [&](const float* xA, const float* xB, const unsigned short* wp,
                    const float* bs, float* outp, int Cin, int Cout, int outCAlloc,
                    int H, int W, int act, int gridY) {
        dim3 g((unsigned)(BATCH * H * (W / 32)), (unsigned)gridY, 1);
        if (Cin == 128) {
            conv3x3_wmma_kernel<128><<<g, dim3(256, 1, 1), SM128, stream>>>(
                xA, xB, wp, bs, outp, Cout, outCAlloc, H, W, act);
        } else {
            conv3x3_wmma_kernel<64><<<g, dim3(256, 1, 1), SM64, stream>>>(
                xA, xB, wp, bs, outp, Cout, outCAlloc, H, W, act);
        }
    };
    auto dcn = [&](const float* x, const float* omp, const unsigned short* wp,
                   const float* bs, float* outp, int H, int W, int act) {
        dim3 g((unsigned)(BATCH * H * (W / 32)), 1, 1);
        dcn_wmma_kernel<<<g, dim3(256, 1, 1), SMDCN, stream>>>(x, omp, wp, bs, outp, H, W, act);
    };
    auto up2 = [&](const float* in, float* outp, int h, int w, float scale) {
        size_t total = (size_t)BATCH * 64 * 4 * h * w;
        int blocks = (int)((total + 255) / 256);
        upsample2x_kernel<<<blocks, 256, 0, stream>>>(in, outp, h, w, scale);
    };

    // ---- pack all weights (every call: deterministic)
    repack(L3_oc_w1, p_L3oc1, 64, 128, 64);
    repack(L3_oc_w2, p_L3oc2, 64, 64, 64);
    repack(L3_dcn_omw, p_L3om, 216, 64, 256);
    repack(L3_dcn_w, p_L3w, 64, 64, 64);
    repack(L2_oc1_w, p_L2oc1, 64, 128, 64);
    repack(L2_oc2_w1, p_L2oc2a, 64, 128, 64);
    repack(L2_oc2_w2, p_L2oc2b, 64, 64, 64);
    repack(L2_dcn_omw, p_L2om, 216, 64, 256);
    repack(L2_dcn_w, p_L2w, 64, 64, 64);
    repack(L2_fea_w, p_L2fea, 64, 128, 64);
    repack(L1_oc1_w, p_L1oc1, 64, 128, 64);
    repack(L1_oc2_w1, p_L1oc2a, 64, 128, 64);
    repack(L1_oc2_w2, p_L1oc2b, 64, 64, 64);
    repack(L1_dcn_omw, p_L1om, 216, 64, 256);
    repack(L1_dcn_w, p_L1w, 64, 64, 64);
    repack(L1_fea_w, p_L1fea, 64, 128, 64);
    repack(cas_oc_w1, p_casoc1, 64, 128, 64);
    repack(cas_oc_w2, p_casoc2, 64, 64, 64);
    repack(cas_dcn_omw, p_casom, 216, 64, 256);
    repack(cas_dcn_w, p_casw, 64, 64, 64);

    // ---- L3 (64x64)
    conv(f1_3, f2_3, p_L3oc1, L3_oc_b1, t3a, 128, 64, 64, 64, 64, 1, 1);
    conv(t3a, nullptr, p_L3oc2, L3_oc_b2, o_l3o, 64, 64, 64, 64, 64, 1, 1);
    conv(o_l3o, nullptr, p_L3om, L3_dcn_omb, om, 64, 216, 216, 64, 64, 0, 4);
    dcn(f1_3, om, p_L3w, L3_dcn_b, l3f, 64, 64, 1);

    // ---- L2 (128x128)
    conv(f1_2, f2_2, p_L2oc1, L2_oc1_b, A2, 128, 64, 64, 128, 128, 1, 1);
    up2(o_l3o, up, 64, 64, 2.0f);
    conv(A2, up, p_L2oc2a, L2_oc2_b1, B2b, 128, 64, 64, 128, 128, 1, 1);
    conv(B2b, nullptr, p_L2oc2b, L2_oc2_b2, o_l2o, 64, 64, 64, 128, 128, 1, 1);
    conv(o_l2o, nullptr, p_L2om, L2_dcn_omb, om, 64, 216, 216, 128, 128, 0, 4);
    dcn(f1_2, om, p_L2w, L2_dcn_b, C2, 128, 128, 0);
    up2(l3f, up, 64, 64, 1.0f);
    conv(C2, up, p_L2fea, L2_fea_b, D2, 128, 64, 64, 128, 128, 1, 1);

    // ---- L1 (256x256)
    conv(f1_1, f2_1, p_L1oc1, L1_oc1_b, A1, 128, 64, 64, 256, 256, 1, 1);
    up2(o_l2o, up, 128, 128, 2.0f);
    conv(A1, up, p_L1oc2a, L1_oc2_b1, B1b, 128, 64, 64, 256, 256, 1, 1);
    conv(B1b, nullptr, p_L1oc2b, L1_oc2_b2, o_l1o, 64, 64, 64, 256, 256, 1, 1);
    conv(o_l1o, nullptr, p_L1om, L1_dcn_omb, om, 64, 216, 216, 256, 256, 0, 4);
    dcn(f1_1, om, p_L1w, L1_dcn_b, C1, 256, 256, 0);
    up2(D2, up, 128, 128, 1.0f);
    conv(C1, up, p_L1fea, L1_fea_b, E1, 128, 64, 64, 256, 256, 0, 1);   // no lrelu

    // ---- cascading refinement (256x256)
    conv(E1, f2_1, p_casoc1, cas_oc_b1, A1, 128, 64, 64, 256, 256, 1, 1);
    conv(A1, nullptr, p_casoc2, cas_oc_b2, o_off, 64, 64, 64, 256, 256, 1, 1);
    conv(o_off, nullptr, p_casom, cas_dcn_omb, om, 64, 216, 216, 256, 256, 0, 4);
    dcn(E1, om, p_casw, cas_dcn_b, o_l1f, 256, 256, 1);
}
